// TransformerBlock_57664230916137
// MI455X (gfx1250) — matrix operand order repaired and checked by reading, not compiled, not run on hardware
//
#include <hip/hip_runtime.h>

// ---- problem constants (match reference) ----
#define Bc   4
#define Sc   2048
#define Dc   1024
#define Hc   16
#define DKc  64
#define DFFc 4096
#define BSc  (Bc * Sc)   // 8192 rows

// ---- vector types (trivial, union-safe) ----
typedef __attribute__((ext_vector_type(16))) _Float16 v16h;
typedef __attribute__((ext_vector_type(8)))  _Float16 v8h;
typedef __attribute__((ext_vector_type(8)))  float    v8f;
typedef __attribute__((ext_vector_type(4)))  float    f4;

union F16x16 { v16h v; f4 f[2]; };
union F16x8  { v8h v; f4 f; _Float16 h[8]; };
union H2Pack { _Float16 h[2]; unsigned int u; };

__device__ __forceinline__ v8f wmma16(v16h a, v16h b, v8f c) {
  // D = A(16x32 f16) * B(32x16 f16) + C(16x16 f32)
  return __builtin_amdgcn_wmma_f32_16x16x32_f16(false, a, false, b, (short)0, c, false, false);
}

__device__ __forceinline__ float gelu_exact(float x) {
  return 0.5f * x * (1.0f + erff(x * 0.70710678118654752f));
}

// ---- CDNA5 async global->LDS DMA (ASYNCcnt path) ----
// LDS byte address = flat addr truncated to 32 bits.
__device__ __forceinline__ void async_copy16(unsigned int lds_addr, const void* gaddr) {
  asm volatile("global_load_async_to_lds_b128 %0, %1, off"
               :: "v"(lds_addr), "v"((unsigned long long)(uintptr_t)gaddr)
               : "memory");
}

__device__ __forceinline__ void wait_async0() {
#if __has_builtin(__builtin_amdgcn_s_wait_asynccnt)
  __builtin_amdgcn_s_wait_asynccnt(0);
#else
  asm volatile("s_wait_asynccnt 0x0" ::: "memory");
#endif
}

// =====================================================================
// fp32 -> f16 conversion, 8 elements/thread, b128 in / b128 out
// =====================================================================
__global__ void cvt_kernel(const float* __restrict__ in, _Float16* __restrict__ out, int n) {
  const int i = (blockIdx.x * blockDim.x + threadIdx.x) * 8;
  if (i >= n) return;
  const f4 a = *(const f4*)(in + i);
  const f4 b = *(const f4*)(in + i + 4);
  F16x8 o;
  o.h[0] = (_Float16)a.x; o.h[1] = (_Float16)a.y;
  o.h[2] = (_Float16)a.z; o.h[3] = (_Float16)a.w;
  o.h[4] = (_Float16)b.x; o.h[5] = (_Float16)b.y;
  o.h[6] = (_Float16)b.z; o.h[7] = (_Float16)b.w;
  *(f4*)(out + i) = o.f;
}

// =====================================================================
// LayerNorm fp32 -> f16 (one 256-thread block per row of D=1024)
// =====================================================================
__global__ __launch_bounds__(256)
void ln_kernel(const float* __restrict__ x, const float* __restrict__ g,
               const float* __restrict__ be, _Float16* __restrict__ out) {
  __shared__ float red[256];
  const int row = blockIdx.x;
  const int tid = threadIdx.x;
  const f4 v = *(const f4*)(x + (size_t)row * Dc + tid * 4);

  red[tid] = v.x + v.y + v.z + v.w;
  __syncthreads();
  for (int o = 128; o > 0; o >>= 1) {
    if (tid < o) red[tid] += red[tid + o];
    __syncthreads();
  }
  const float mu = red[0] * (1.0f / Dc);
  __syncthreads();

  const float d0 = v.x - mu, d1 = v.y - mu, d2 = v.z - mu, d3 = v.w - mu;
  red[tid] = d0 * d0 + d1 * d1 + d2 * d2 + d3 * d3;
  __syncthreads();
  for (int o = 128; o > 0; o >>= 1) {
    if (tid < o) red[tid] += red[tid + o];
    __syncthreads();
  }
  const float var = red[0] * (1.0f / Dc);
  const float rs = rsqrtf(var + 1e-5f);

  const int c = tid * 4;
  _Float16* op = out + (size_t)row * Dc + c;
  op[0] = (_Float16)(d0 * rs * g[c + 0] + be[c + 0]);
  op[1] = (_Float16)(d1 * rs * g[c + 1] + be[c + 1]);
  op[2] = (_Float16)(d2 * rs * g[c + 2] + be[c + 2]);
  op[3] = (_Float16)(d3 * rs * g[c + 3] + be[c + 3]);
}

// =====================================================================
// f16 GEMM: C[M,N] = A[M,K] @ W[K,N]  (+bias / GELU / residual, templated)
// 128x128 block tile, BK=32, 256 thr = 8 waves (wave tile 32x64).
// Double-buffered LDS; A tile via async global->LDS DMA, W tile via
// registers with b32-packed transpose stores. Pipeline overlaps next-tile
// DMA with the current tile's 8 WMMAs.
// =====================================================================
__device__ __forceinline__ void gemm_loadB_regs(const _Float16* __restrict__ W, int N, int bN,
                                                int pr, int n0, int k0, F16x8& r0, F16x8& r1) {
  const _Float16* s0 = W + (size_t)(k0 + 2 * pr) * N + bN + n0;
  r0.f = *(const f4*)s0;
  r1.f = *(const f4*)(s0 + N);
}
__device__ __forceinline__ void gemm_storeB(_Float16 (*bs)[40], int pr, int n0,
                                            const F16x8& r0, const F16x8& r1) {
#pragma unroll
  for (int j = 0; j < 8; ++j) {
    H2Pack pk;
    pk.h[0] = r0.h[j];
    pk.h[1] = r1.h[j];
    *(unsigned int*)&bs[n0 + j][2 * pr] = pk.u;
  }
}
__device__ __forceinline__ void gemm_asyncA(const _Float16* __restrict__ A, int K, int bM,
                                            int ar, int ac, int k0, _Float16 (*as)[40]) {
  const _Float16* src = A + (size_t)(bM + ar) * K + k0 + ac;
  const unsigned int l = (unsigned int)(uintptr_t)&as[ar][ac];
  async_copy16(l, src);
  async_copy16(l + 16, src + 8);
}

template <bool GELU, bool HASB, bool HASR, bool OUTF, bool OUTH>
__global__ __launch_bounds__(256)
void gemm_kernel(const _Float16* __restrict__ A, const _Float16* __restrict__ W,
                 const float* __restrict__ bias, const float* __restrict__ resid,
                 float* __restrict__ outF, _Float16* __restrict__ outH,
                 int M, int N, int K) {
  __shared__ __attribute__((aligned(16))) _Float16 As[2][128][40]; // [m][k]
  __shared__ __attribute__((aligned(16))) _Float16 Bs[2][128][40]; // [n][k] (W transposed)

  const int tid  = threadIdx.x;
  const int w    = tid >> 5;
  const int lane = tid & 31;
  const int hx   = lane >> 4;
  const int l15  = lane & 15;
  const int wm   = w >> 1;  // 0..3 : 32-row slice
  const int wn   = w & 1;   // 0..1 : 64-col slice
  const int bM   = blockIdx.y * 128;
  const int bN   = blockIdx.x * 128;

  const int ar = tid >> 1, ac = (tid & 1) * 16;  // A loader role
  const int pr = tid >> 4, n0 = (tid & 15) * 8;  // B loader role

  v8f acc[2][4];
  const v8f zero = {};
#pragma unroll
  for (int t = 0; t < 2; ++t)
#pragma unroll
    for (int n = 0; n < 4; ++n) acc[t][n] = zero;

  // ---- prologue: stage tile 0 ----
  F16x8 r0, r1;
  gemm_asyncA(A, K, bM, ar, ac, 0, As[0]);
  gemm_loadB_regs(W, N, bN, pr, n0, 0, r0, r1);
  gemm_storeB(Bs[0], pr, n0, r0, r1);
  wait_async0();
  __syncthreads();

  int buf = 0;
  for (int k0 = 0; k0 < K; k0 += 32) {
    const bool hasNext = (k0 + 32) < K;
    if (hasNext) {  // kick next tile's DMA + W loads before computing
      gemm_asyncA(A, K, bM, ar, ac, k0 + 32, As[buf ^ 1]);
      gemm_loadB_regs(W, N, bN, pr, n0, k0 + 32, r0, r1);
    }

    const _Float16(*as)[40] = As[buf];
    const _Float16(*bs)[40] = Bs[buf];
    F16x16 af[2];
#pragma unroll
    for (int t = 0; t < 2; ++t) {
      const _Float16* p = &as[wm * 32 + t * 16 + l15][hx * 16];
      af[t].f[0] = *(const f4*)p;         // K {0..7} / {16..23}
      af[t].f[1] = *(const f4*)(p + 8);   // K {8..15} / {24..31}
    }
#pragma unroll
    for (int n = 0; n < 4; ++n) {
      F16x16 bf;
      const _Float16* p = &bs[wn * 64 + n * 16 + l15][hx * 16];
      bf.f[0] = *(const f4*)p;            // 16 consecutive K at fixed N
      bf.f[1] = *(const f4*)(p + 8);
#pragma unroll
      for (int t = 0; t < 2; ++t)
        acc[t][n] = wmma16(af[t].v, bf.v, acc[t][n]);
    }

    if (hasNext) {
      gemm_storeB(Bs[buf ^ 1], pr, n0, r0, r1);
      wait_async0();
      __syncthreads();
      buf ^= 1;
    }
  }

  // ---- straight-line epilogue (C layout: m = i + 8*hx, n = l15) ----
#pragma unroll
  for (int t = 0; t < 2; ++t) {
#pragma unroll
    for (int n = 0; n < 4; ++n) {
      const int col = bN + wn * 64 + n * 16 + l15;
      float bv = 0.0f;
      if constexpr (HASB) bv = bias[col];
#pragma unroll
      for (int i = 0; i < 8; ++i) {
        const int row = bM + wm * 32 + t * 16 + i + 8 * hx;
        float val = acc[t][n][i] + bv;
        if constexpr (GELU) val = gelu_exact(val);
        if constexpr (HASR) val += resid[(size_t)row * N + col];
        if constexpr (OUTF) outF[(size_t)row * N + col] = val;
        if constexpr (OUTH) outH[(size_t)row * N + col] = (_Float16)val;
      }
    }
  }
}

// =====================================================================
// causal flash attention: grid (S/64, B*H), 128 threads = 4 waves,
// each wave owns a 16-row Q tile; K/V tiles (64 rows) shared in LDS.
// K tile arrives via async global->LDS DMA; V is transposed through
// registers with b32-packed stores; next tiles are prefetched.
// =====================================================================
__global__ __launch_bounds__(128)
void attn_kernel(const _Float16* __restrict__ Q, const _Float16* __restrict__ K,
                 const _Float16* __restrict__ V, _Float16* __restrict__ O) {
  __shared__ __attribute__((aligned(16))) _Float16 Kt[64][72];     // [krow][dk]
  __shared__ __attribute__((aligned(16))) _Float16 Vt[64][72];     // [dk][krow] (transposed)
  __shared__ __attribute__((aligned(16))) _Float16 Pb[4][16][72];  // per-wave P staging

  const int qblk = blockIdx.x;
  const int bh   = blockIdx.y;
  const int b    = bh / Hc;
  const int h    = bh % Hc;
  const int tid  = threadIdx.x;
  const int w    = tid >> 5;
  const int lane = tid & 31;
  const int hx   = lane >> 4;
  const int l15  = lane & 15;

  const int q0 = qblk * 64 + w * 16;
  const size_t baseBH = ((size_t)b * Sc) * Dc + (size_t)h * DKc;

  // Q fragments (A layout), rows q0+l15, dk split {0..31} / {32..63}
  F16x16 qa[2];
  {
    const _Float16* qrow = Q + baseBH + (size_t)(q0 + l15) * Dc;
#pragma unroll
    for (int f = 0; f < 2; ++f) {
      const _Float16* p = qrow + f * 32 + hx * 16;
      qa[f].f[0] = *(const f4*)p;
      qa[f].f[1] = *(const f4*)(p + 8);
    }
  }

  float m_r[8], l_r[8];
  v8f oacc[4];
  const v8f zero = {};
#pragma unroll
  for (int i = 0; i < 8; ++i) { m_r[i] = -1e30f; l_r[i] = 0.0f; }
#pragma unroll
  for (int n = 0; n < 4; ++n) oacc[n] = zero;

  const int kend = (qblk + 1) * 64;
  for (int k0 = 0; k0 < kend; k0 += 64) {
    __syncthreads();  // previous tile's readers are done
    // ---- K tile: async DMA straight into LDS (natural layout) ----
    {
      int idx = tid;  // 64 rows * 8 chunks of 8 halves = 512 tasks
#pragma unroll
      for (int it = 0; it < 4; ++it, idx += 128) {
        const int r = idx >> 3;
        const int c = (idx & 7) * 8;
        async_copy16((unsigned int)(uintptr_t)&Kt[r][c],
                     K + baseBH + (size_t)(k0 + r) * Dc + c);
      }
    }
    // ---- V tile: transpose via regs, b32-packed row-pair stores ----
    {
      int idx = tid;  // 32 row-pairs * 8 chunks = 256 tasks
#pragma unroll
      for (int it = 0; it < 2; ++it, idx += 128) {
        const int prr = idx >> 3;
        const int c   = (idx & 7) * 8;
        const _Float16* s0 = V + baseBH + (size_t)(k0 + 2 * prr) * Dc + c;
        F16x8 v0, v1;
        v0.f = *(const f4*)s0;
        v1.f = *(const f4*)(s0 + Dc);
#pragma unroll
        for (int j = 0; j < 8; ++j) {
          H2Pack pk;
          pk.h[0] = v0.h[j];
          pk.h[1] = v1.h[j];
          *(unsigned int*)&Vt[c + j][2 * prr] = pk.u;
        }
      }
    }
    // prefetch next tiles while this one finishes (global_prefetch_b8)
    if (k0 + 64 < kend) {
      __builtin_prefetch(K + baseBH + (size_t)(k0 + 64 + (tid >> 1)) * Dc, 0, 1);
      __builtin_prefetch(V + baseBH + (size_t)(k0 + 64 + (tid >> 1)) * Dc, 0, 1);
    }
    wait_async0();
    __syncthreads();

    // ---- scores: 4 tiles of 16x16, each = 2 WMMAs over dk ----
    v8f sacc[4];
#pragma unroll
    for (int kn = 0; kn < 4; ++kn) {
      v8f s = zero;
#pragma unroll
      for (int f = 0; f < 2; ++f) {
        F16x16 bf;
        const _Float16* p = &Kt[kn * 16 + l15][f * 32 + hx * 16];
        bf.f[0] = *(const f4*)p;
        bf.f[1] = *(const f4*)(p + 8);
        s = wmma16(qa[f].v, bf.v, s);
      }
      sacc[kn] = s;
    }

    // ---- causal mask + scale + online softmax (C layout: m=i+8*hx, n=l15) ----
#pragma unroll
    for (int i = 0; i < 8; ++i) {
      const int qr = q0 + i + 8 * hx;
      float mx = -1e30f;
#pragma unroll
      for (int kn = 0; kn < 4; ++kn) {
        const int kc = k0 + kn * 16 + l15;
        float v = sacc[kn][i] * 0.125f;  // 1/sqrt(DK)
        if (kc > qr) v = -1e30f;
        sacc[kn][i] = v;
        mx = fmaxf(mx, v);
      }
#pragma unroll
      for (int msk = 1; msk < 16; msk <<= 1) mx = fmaxf(mx, __shfl_xor(mx, msk, 32));
      const float mnew = fmaxf(m_r[i], mx);
      const float alpha = __expf(m_r[i] - mnew);
      l_r[i] *= alpha;
#pragma unroll
      for (int n = 0; n < 4; ++n) oacc[n][i] *= alpha;
      float rsum = 0.0f;
#pragma unroll
      for (int kn = 0; kn < 4; ++kn) {
        const float p = __expf(sacc[kn][i] - mnew);
        sacc[kn][i] = p;
        rsum += p;
      }
#pragma unroll
      for (int msk = 1; msk < 16; msk <<= 1) rsum += __shfl_xor(rsum, msk, 32);
      l_r[i] += rsum;
      m_r[i] = mnew;
    }

    // ---- P: C layout -> LDS -> A fragments (wave-local) ----
#pragma unroll
    for (int kn = 0; kn < 4; ++kn)
#pragma unroll
      for (int i = 0; i < 8; ++i)
        Pb[w][i + 8 * hx][kn * 16 + l15] = (_Float16)sacc[kn][i];

    F16x16 pa[2];
#pragma unroll
    for (int c = 0; c < 2; ++c) {
      const _Float16* p = &Pb[w][l15][c * 32 + hx * 16];
      pa[c].f[0] = *(const f4*)p;
      pa[c].f[1] = *(const f4*)(p + 8);
    }

    // ---- O += P @ V : 4 output tiles x 2 k-chunks ----
#pragma unroll
    for (int n = 0; n < 4; ++n) {
#pragma unroll
      for (int c = 0; c < 2; ++c) {
        F16x16 bf;
        const _Float16* p = &Vt[n * 16 + l15][c * 32 + hx * 16];
        bf.f[0] = *(const f4*)p;
        bf.f[1] = *(const f4*)(p + 8);
        oacc[n] = wmma16(pa[c].v, bf.v, oacc[n]);
      }
    }
  }

  // ---- normalize and store O (f16, [B,S,D] with head offset) ----
#pragma unroll
  for (int n = 0; n < 4; ++n)
#pragma unroll
    for (int i = 0; i < 8; ++i) {
      const int qr = q0 + i + 8 * hx;
      const float val = oacc[n][i] / l_r[i];
      O[baseBH + (size_t)qr * Dc + n * 16 + l15] = (_Float16)val;
    }
}

// =====================================================================
// driver
// =====================================================================
extern "C" void kernel_launch(void* const* d_in, const int* in_sizes, int n_in,
                              void* d_out, int out_size, void* d_ws, size_t ws_size,
                              hipStream_t stream) {
  (void)in_sizes; (void)n_in; (void)out_size; (void)ws_size;
  const float* x   = (const float*)d_in[0];
  const float* wq  = (const float*)d_in[1];
  const float* wk  = (const float*)d_in[2];
  const float* wv  = (const float*)d_in[3];
  const float* wo  = (const float*)d_in[4];
  const float* bo  = (const float*)d_in[5];
  const float* w1  = (const float*)d_in[6];
  const float* b1  = (const float*)d_in[7];
  const float* w2  = (const float*)d_in[8];
  const float* b2  = (const float*)d_in[9];
  const float* g1  = (const float*)d_in[10];
  const float* be1 = (const float*)d_in[11];
  const float* g2  = (const float*)d_in[12];
  const float* be2 = (const float*)d_in[13];
  float* out = (float*)d_out;
  char* ws = (char*)d_ws;

  const size_t MB = 1ull << 20;
  _Float16* wq16 = (_Float16*)(ws + 0 * MB);    // 2 MB
  _Float16* wk16 = (_Float16*)(ws + 2 * MB);    // 2 MB
  _Float16* wv16 = (_Float16*)(ws + 4 * MB);    // 2 MB
  _Float16* wo16 = (_Float16*)(ws + 6 * MB);    // 2 MB
  _Float16* w116 = (_Float16*)(ws + 8 * MB);    // 8 MB
  _Float16* w216 = (_Float16*)(ws + 16 * MB);   // 8 MB
  _Float16* h1   = (_Float16*)(ws + 24 * MB);   // 16 MB (reused as h2)
  _Float16* Qb   = (_Float16*)(ws + 40 * MB);   // 16 MB
  _Float16* Kb   = (_Float16*)(ws + 56 * MB);   // 16 MB
  _Float16* Vb   = (_Float16*)(ws + 72 * MB);   // 16 MB
  _Float16* Ob   = (_Float16*)(ws + 88 * MB);   // 16 MB
  float*    x2   = (float*)   (ws + 104 * MB);  // 32 MB
  _Float16* h2   = (_Float16*)(ws + 24 * MB);   // aliases h1 (dead by then)
  _Float16* act  = (_Float16*)(ws + 40 * MB);   // 64 MB, aliases Q/K/V/O (dead by then)

  const int nDD = Dc * Dc;
  const int nDF = Dc * DFFc;
  cvt_kernel<<<nDD / (8 * 256), 256, 0, stream>>>(wq, wq16, nDD);
  cvt_kernel<<<nDD / (8 * 256), 256, 0, stream>>>(wk, wk16, nDD);
  cvt_kernel<<<nDD / (8 * 256), 256, 0, stream>>>(wv, wv16, nDD);
  cvt_kernel<<<nDD / (8 * 256), 256, 0, stream>>>(wo, wo16, nDD);
  cvt_kernel<<<nDF / (8 * 256), 256, 0, stream>>>(w1, w116, nDF);
  cvt_kernel<<<nDF / (8 * 256), 256, 0, stream>>>(w2, w216, nDF);

  // LN1 -> h1
  ln_kernel<<<BSc, 256, 0, stream>>>(x, g1, be1, h1);

  // QKV projections (f16 out only)
  const dim3 blk(256);
  const dim3 gD(Dc / 128, BSc / 128);  // (8, 64)
  gemm_kernel<false, false, false, false, true><<<gD, blk, 0, stream>>>(
      h1, wq16, nullptr, nullptr, nullptr, Qb, BSc, Dc, Dc);
  gemm_kernel<false, false, false, false, true><<<gD, blk, 0, stream>>>(
      h1, wk16, nullptr, nullptr, nullptr, Kb, BSc, Dc, Dc);
  gemm_kernel<false, false, false, false, true><<<gD, blk, 0, stream>>>(
      h1, wv16, nullptr, nullptr, nullptr, Vb, BSc, Dc, Dc);

  // causal attention
  attn_kernel<<<dim3(Sc / 64, Bc * Hc), dim3(128), 0, stream>>>(Qb, Kb, Vb, Ob);

  // output projection + bias + residual -> x2 (fp32)
  gemm_kernel<false, true, true, true, false><<<gD, blk, 0, stream>>>(
      Ob, wo16, bo, x, x2, nullptr, BSc, Dc, Dc);

  // LN2 -> h2
  ln_kernel<<<BSc, 256, 0, stream>>>(x2, g2, be2, h2);

  // FFN up (bias + exact GELU fused) -> act (f16)
  gemm_kernel<true, true, false, false, true><<<dim3(DFFc / 128, BSc / 128), blk, 0, stream>>>(
      h2, w116, b1, nullptr, nullptr, act, BSc, DFFc, Dc);

  // FFN down + bias + residual -> out (fp32)
  gemm_kernel<false, true, true, true, false><<<gD, blk, 0, stream>>>(
      act, w216, b2, x2, out, nullptr, BSc, Dc, DFFc);
}
